// Edgewise_74328704024938
// MI455X (gfx1250) — matrix operand order repaired and checked by reading, not compiled, not run on hardware
//
#include <hip/hip_runtime.h>
#include <hip/hip_bf16.h>

typedef __bf16 bf16_t;
typedef __attribute__((ext_vector_type(16))) __bf16 v16bf;
typedef __attribute__((ext_vector_type(8)))  float  v8f;
typedef __attribute__((ext_vector_type(4))) unsigned tdm_v4u;
typedef __attribute__((ext_vector_type(8))) int      tdm_v8i;
typedef __attribute__((ext_vector_type(4))) int      tdm_v4i;

#if defined(__HIP_DEVICE_COMPILE__) && defined(__has_builtin)
#if __has_builtin(__builtin_amdgcn_tensor_load_to_lds) && \
    __has_builtin(__builtin_amdgcn_s_wait_tensorcnt)
#define HAVE_TDM 1
#endif
#endif
#ifndef HAVE_TDM
#define HAVE_TDM 0
#endif

#define NUM_E 65536
#define NUM_N 8192

// ---- packed bf16 weight layout inside workspace (element offsets) ----
// All GEMM weights are stored FRAGMENT-PACKED: for weight W[Ka x Nw],
// element ((kc*(Nw/16)+nt)*32 + lane)*16 + i  ==  W[kc*32 + (lane>>4)*16 + i][nt*16 + (lane&15)]
#define OFF_RW1   0L          // rad_w1   [128 x 64]        8192
#define OFF_RW2   8192L       // rad_w2   [64 x 2304]       147456
#define OFF_W0C1  155648L     // conv1_w0 [768 x 640]       491520
#define OFF_W1C1  647168L     // conv1 m1 [1024 x 512]      524288
#define OFF_W2C1  1171456L    // conv1 m2 [512 x 256]       131072
#define OFF_W0C2  1302528L    // conv2_w0 [384 x 384]       147456
#define OFF_W1C2  1449984L    // conv2 m1 [512 x 512]       262144
#define OFF_W2C2  1712128L    // conv2 m2 [256 x 256]       65536
#define TOTAL_W   1777664L

// byte offsets of big buffers in workspace
#define B0_OFF 3555328L                         // bufA  : [E,2304] bf16
#define B1_OFF (B0_OFF + 301989888L)            // bufY1 : [E,1408] bf16
#define B2_OFF (B1_OFF + 184549376L)            // bufA2 : [E,1152] bf16
#define B3_OFF (B2_OFF + 150994944L)            // bufY2 : [E,1152] bf16

// ---------------- WMMA helpers (CDNA5 16x16x32 bf16) ----------------
__device__ inline v8f wmma_bf16(v16bf a, v16bf b, v8f c) {
  return __builtin_amdgcn_wmma_f32_16x16x32_bf16(
      false, a, false, b, (short)0, c, false, false);
}

union AFragU { v16bf v; uint4 q[2]; };

// A fragment from a row-major 16x32 tile (leading dim lda, 16B-aligned rows):
// lane holds row m=lane&15, one contiguous 16-element run at K=(lane>>4)*16.
__device__ inline v16bf load_a_frag(const bf16_t* tileBase, int lda, int lane) {
  const bf16_t* p = tileBase + (lane & 15) * lda + ((lane >> 4) * 16);
  AFragU f;
  f.q[0] = *(const uint4*)(p);
  f.q[1] = *(const uint4*)(p + 8);
  return f.v;
}

// B fragment from fragment-packed weights: one aligned 32-byte load.
__device__ inline v16bf load_b_frag_packed(const bf16_t* p) {
  return *(const v16bf*)p;
}

// fast activations (results feed bf16 storage; v_rcp_f32 precision is plenty)
__device__ inline float sigmoidf_(float x) {
  return __builtin_amdgcn_rcpf(1.f + __expf(-x));
}
__device__ inline float siluf_(float x) { return x * sigmoidf_(x); }

#if HAVE_TDM
// Tensor Data Mover: 2D tile (tileW x tileH, bf16) global -> compact row-major LDS.
// D# per CDNA5 ISA (group0 + group1; groups 2/3 zero -> 2D).
// This toolchain's builtin arity is 6: (v4u, v8i, v4i, v4i, v8i, i32 cpol).
__device__ inline void tdm_load_2d(unsigned ldsOff, const void* gp,
                                   unsigned tileW, unsigned tileH, unsigned strideElems) {
  unsigned long long ga = (unsigned long long)(uintptr_t)gp;
  tdm_v4u g0 = { 1u,                                   // count=1 (valid user D#)
                 ldsOff,                               // lds_addr (bytes)
                 (unsigned)ga,                         // global_addr[31:0]
                 (unsigned)((ga >> 32) & 0x01ffffffu) | 0x80000000u };  // addr[56:32] | type=2
  tdm_v8i g1 = { (int)(1u << 16),                      // data_size=1 (2 bytes), wg_mask=0
                 (int)((tileW & 0xffffu) << 16),                                   // tensor_dim0[15:0]
                 (int)(((tileW >> 16) & 0xffffu) | ((tileH & 0xffffu) << 16)),     // dim0 hi | dim1 lo
                 (int)(((tileH >> 16) & 0xffffu) | ((tileW & 0xffffu) << 16)),     // dim1 hi | tile_dim0
                 (int)(tileH & 0xffffu),               // tile_dim1 (tile_dim2=0 -> 2D)
                 (int)strideElems,                     // tensor_dim0_stride[31:0]
                 0, 0 };
  tdm_v4i g2 = { 0, 0, 0, 0 };
  tdm_v4i g3 = { 0, 0, 0, 0 };
  tdm_v8i gx = { 0, 0, 0, 0, 0, 0, 0, 0 };
  __builtin_amdgcn_tensor_load_to_lds(g0, g1, g2, g3, gx, 0);
}
#endif

// ---------------- kernel: zero output ----------------
__global__ void zero_kernel(float* __restrict__ p, long n) {
  long i = (long)blockIdx.x * blockDim.x + threadIdx.x;
  if (i < n) p[i] = 0.f;
}

// ---------------- kernel: pack weights to bf16, fragment-ordered ----------------
__global__ void pack_kernel(const float* __restrict__ rw1, const float* __restrict__ rw2,
                            const float* __restrict__ c1w0,
                            const float* __restrict__ c1wr1, const float* __restrict__ c1wi1,
                            const float* __restrict__ c1wr2, const float* __restrict__ c1wi2,
                            const float* __restrict__ c2w0,
                            const float* __restrict__ c2wr1, const float* __restrict__ c2wi1,
                            const float* __restrict__ c2wr2, const float* __restrict__ c2wi2,
                            bf16_t* __restrict__ wb) {
  for (long i = (long)blockIdx.x * blockDim.x + threadIdx.x; i < TOTAL_W;
       i += (long)gridDim.x * blockDim.x) {
    long off; int Nw, region;
    if      (i < OFF_RW2)  { off = OFF_RW1;  Nw = 64;   region = 0; }
    else if (i < OFF_W0C1) { off = OFF_RW2;  Nw = 2304; region = 1; }
    else if (i < OFF_W1C1) { off = OFF_W0C1; Nw = 640;  region = 2; }
    else if (i < OFF_W2C1) { off = OFF_W1C1; Nw = 512;  region = 3; }
    else if (i < OFF_W0C2) { off = OFF_W2C1; Nw = 256;  region = 4; }
    else if (i < OFF_W1C2) { off = OFF_W0C2; Nw = 384;  region = 5; }
    else if (i < OFF_W2C2) { off = OFF_W1C2; Nw = 512;  region = 6; }
    else                   { off = OFF_W2C2; Nw = 256;  region = 7; }
    long j = i - off;
    int lane = (int)((j >> 4) & 31);
    int fi   = (int)(j & 15);
    long tile = j >> 9;
    int ntn = Nw >> 4;
    int kc = (int)(tile / ntn), nt = (int)(tile - (long)kc * ntn);
    int k = kc * 32 + (lane >> 4) * 16 + fi;
    int n = nt * 16 + (lane & 15);
    float v;
    switch (region) {
      case 0: v = rw1[k * 64 + n]; break;
      case 1: v = rw2[k * 2304 + n]; break;
      case 2: v = c1w0[k * 640 + n]; break;
      case 3: v = (k < 512) ? ((n < 256) ? c1wr1[k * 256 + n] : c1wi1[k * 256 + n - 256])
                            : ((n < 256) ? -c1wi1[(k - 512) * 256 + n]
                                         : c1wr1[(k - 512) * 256 + n - 256]); break;
      case 4: v = (k < 256) ? ((n < 128) ? c1wr2[k * 128 + n] : c1wi2[k * 128 + n - 128])
                            : ((n < 128) ? -c1wi2[(k - 256) * 128 + n]
                                         : c1wr2[(k - 256) * 128 + n - 128]); break;
      case 5: v = c2w0[k * 384 + n]; break;
      case 6: v = (k < 256) ? ((n < 256) ? c2wr1[k * 256 + n] : c2wi1[k * 256 + n - 256])
                            : ((n < 256) ? -c2wi1[(k - 256) * 256 + n]
                                         : c2wr1[(k - 256) * 256 + n - 256]); break;
      default: v = (k < 128) ? ((n < 128) ? c2wr2[k * 128 + n] : c2wi2[k * 128 + n - 128])
                             : ((n < 128) ? -c2wi2[(k - 128) * 128 + n]
                                          : c2wr2[(k - 128) * 128 + n - 128]); break;
    }
    wb[i] = (bf16_t)v;
  }
}

// ---------------- kernel: prep (gather + radial MLP via WMMA + Wigner rotate) ----
__global__ __launch_bounds__(128) void prep_kernel(
    const float* __restrict__ node_feats, const float* __restrict__ edge_embeds,
    const int* __restrict__ senders, const int* __restrict__ receivers,
    const float* __restrict__ wigner,
    const bf16_t* __restrict__ wb, const float* __restrict__ rad_b1,
    const float* __restrict__ rad_b2, bf16_t* __restrict__ bufA) {
  __shared__ bf16_t feats[16][1152];   // gathered node feats (one half)
  __shared__ float  wig[16][81];
  __shared__ bf16_t emb[16][128];
  __shared__ bf16_t s1[16][64];        // silu(layer1)
  __shared__ float  radc[16][128];     // radial chunk for current (k, half)

  const int t = threadIdx.x;
  const int wv = t >> 5, lane = t & 31;
  const long e0 = (long)blockIdx.x * 16;
  const int half = blockIdx.y;
  const int* idxArr = half ? receivers : senders;
  const bf16_t* radW1p = wb + OFF_RW1;
  const bf16_t* radW2p = wb + OFF_RW2;

  for (int i = t; i < 16 * 128; i += 128)
    emb[i >> 7][i & 127] = (bf16_t)edge_embeds[(e0 + (i >> 7)) * 128 + (i & 127)];
  for (int i = t; i < 16 * 81; i += 128)
    wig[i / 81][i % 81] = wigner[(e0 + i / 81) * 81 + (i % 81)];
  for (int i = t; i < 16 * 1152; i += 128) {
    int le = i / 1152, cc = i % 1152;
    int node = idxArr[e0 + le];
    feats[le][cc] = (bf16_t)node_feats[(long)node * 1152 + cc];
  }
  __syncthreads();

  // layer1: s1 = silu(emb @ radW1 + b1)  [16 x 64]; each wave owns one 16-col tile
  {
    v8f acc = {};
#pragma unroll
    for (int kc = 0; kc < 4; ++kc) {
      v16bf a = load_a_frag(&emb[0][kc * 32], 128, lane);
      v16bf b = load_b_frag_packed(radW1p + ((long)(kc * 4 + wv) * 32 + lane) * 16);
      acc = wmma_bf16(a, b, acc);
    }
    int n = lane & 15, mb = (lane >> 4) * 8;
#pragma unroll
    for (int v = 0; v < 8; ++v) {
      float x = acc[v] + rad_b1[wv * 16 + n];
      s1[mb + v][wv * 16 + n] = (bf16_t)siluf_(x);
    }
  }
  __syncthreads();

  const int permPos[9] = {0, 5, 1, 3, 8, 6, 2, 4, 7};  // row k -> packed position
  for (int k = 0; k < 9; ++k) {
#pragma unroll
    for (int sub = 0; sub < 2; ++sub) {
      int ntl = wv * 2 + sub;
      int colg = k * 256 + half * 128 + ntl * 16;
      int ntg = colg >> 4;                 // global N-tile index in radW2 (ntn=144)
      v8f acc = {};
#pragma unroll
      for (int kc = 0; kc < 2; ++kc) {
        v16bf a = load_a_frag(&s1[0][kc * 32], 64, lane);
        v16bf b = load_b_frag_packed(radW2p + ((long)(kc * 144 + ntg) * 32 + lane) * 16);
        acc = wmma_bf16(a, b, acc);
      }
      int n = lane & 15, mb = (lane >> 4) * 8;
#pragma unroll
      for (int v = 0; v < 8; ++v) radc[mb + v][ntl * 16 + n] = acc[v] + rad_b2[colg + n];
    }
    __syncthreads();
    for (int i = t; i < 16 * 128; i += 128) {
      int le = i >> 7, c = i & 127;
      float m = 0.f;
#pragma unroll
      for (int j = 0; j < 9; ++j) m += wig[le][k * 9 + j] * (float)feats[le][j * 128 + c];
      float val = m * radc[le][c];
      bufA[(e0 + le) * 2304 + permPos[k] * 256 + half * 128 + c] = (bf16_t)val;
    }
    __syncthreads();
  }
}

// ---------------- kernel: WMMA GEMM, TDM-staged A tile, frag-packed B ----------------
// Y[m, yColOff+n] = sum_k A[m, aColOff+k] * Wp[k, n] (+ bias[n]); bf16 in/out, f32 acc.
// Block: 256 threads (8 waves) -> 128(M) x 64(N) tile; K stepped by 32.
__global__ __launch_bounds__(256) void gemm_bf16_kernel(
    const bf16_t* __restrict__ A, long ldA, long aColOff,
    const bf16_t* __restrict__ Wp, int Nw, const float* __restrict__ bias,
    bf16_t* __restrict__ Y, long ldY, long yColOff, int Ka) {
  __shared__ bf16_t At[2][128][32];
  const int t = threadIdx.x;
  const int wv = t >> 5, lane = t & 31;
  const long mBase = (long)blockIdx.x * 128;
  const int nBase = blockIdx.y * 64;
  const int ntn = Nw >> 4;
  const int nt0 = nBase >> 4;
  v8f acc[4] = {};
  const int nk = Ka >> 5;

#if HAVE_TDM
  if (t == 0)
    tdm_load_2d((unsigned)(uintptr_t)&At[0][0][0],
                A + mBase * ldA + aColOff, 32, 128, (unsigned)ldA);
#endif
  for (int kc = 0; kc < nk; ++kc) {
    bf16_t(*Ab)[32] = At[kc & 1];
#if HAVE_TDM
    if (t == 0 && kc + 1 < nk)
      tdm_load_2d((unsigned)(uintptr_t)&At[(kc + 1) & 1][0][0],
                  A + mBase * ldA + aColOff + (kc + 1) * 32, 32, 128, (unsigned)ldA);
    if (t == 0) {
      if (kc + 1 < nk) __builtin_amdgcn_s_wait_tensorcnt((short)1);
      else             __builtin_amdgcn_s_wait_tensorcnt((short)0);
    }
#else
#pragma unroll
    for (int i = 0; i < 16; ++i) {
      int lin = t + 256 * i;
      int r = lin >> 5, c = lin & 31;
      Ab[r][c] = A[(mBase + r) * ldA + aColOff + kc * 32 + c];
    }
#endif
    __syncthreads();
    v16bf af = load_a_frag(&Ab[wv * 16][0], 32, lane);
    const bf16_t* bp = Wp + ((long)(kc * ntn + nt0) * 32 + lane) * 16;
    if (kc + 1 < nk)
      __builtin_prefetch(Wp + ((long)((kc + 1) * ntn + nt0) * 32 + lane) * 16, 0, 2);
#pragma unroll
    for (int nt = 0; nt < 4; ++nt) {
      v16bf bfg = load_b_frag_packed(bp + (long)nt * 512);
      acc[nt] = wmma_bf16(af, bfg, acc[nt]);
    }
    __syncthreads();
  }
  const int n = lane & 15, mb = (lane >> 4) * 8;
#pragma unroll
  for (int nt = 0; nt < 4; ++nt) {
    int col = nBase + nt * 16 + n;
    float bv = bias ? bias[col] : 0.f;
#pragma unroll
    for (int v = 0; v < 8; ++v) {
      long row = mBase + wv * 16 + mb + v;
      Y[row * ldY + yColOff + col] = (bf16_t)(acc[nt][v] + bv);
    }
  }
}

// ---------------- kernel: gated activation + repack conv2 input ----------------
__global__ __launch_bounds__(128) void gate_kernel(const bf16_t* __restrict__ Y1,
                                                   bf16_t* __restrict__ A2) {
  const long e = blockIdx.x;
  const int c = threadIdx.x;
  const bf16_t* y = Y1 + e * 1408;
  bf16_t* o = A2 + e * 1152;
  const float g0 = sigmoidf_((float)y[384 + c]);
  const float g1 = sigmoidf_((float)y[512 + c]);
  o[c]        = (bf16_t)siluf_((float)y[c]);            // row0
  o[128 + c]  = (bf16_t)((float)y[128 + c]  * g0);      // row2 (l=1)
  o[256 + c]  = (bf16_t)((float)y[256 + c]  * g1);      // row6 (l=2)
  o[384 + c]  = (bf16_t)((float)y[640 + c]  * g0);      // row3
  o[512 + c]  = (bf16_t)((float)y[768 + c]  * g1);      // row7
  o[640 + c]  = (bf16_t)((float)y[896 + c]  * g0);      // row1
  o[768 + c]  = (bf16_t)((float)y[1024 + c] * g1);      // row5
  o[896 + c]  = (bf16_t)((float)y[1152 + c] * g1);      // row8
  o[1024 + c] = (bf16_t)((float)y[1280 + c] * g1);      // row4
}

// ---------------- kernel: envelope * Wigner^T rotate-back + scatter-add ----------
__global__ __launch_bounds__(128) void scatter_kernel(
    const bf16_t* __restrict__ Y2, const float* __restrict__ wigner,
    const float* __restrict__ envelope, const int* __restrict__ receivers,
    float* __restrict__ out) {
  __shared__ float ys[9][128];
  __shared__ float wg[81];
  const long e = blockIdx.x;
  const int c = threadIdx.x;
  const int rowOf[9] = {0, 2, 6, 3, 7, 1, 5, 8, 4};
#pragma unroll
  for (int p = 0; p < 9; ++p) ys[p][c] = (float)Y2[e * 1152 + p * 128 + c];
  if (c < 81) wg[c] = wigner[e * 81 + c];
  __syncthreads();
  const float env = envelope[e];
  const int r = receivers[e];
  float* ob = out + (long)r * 1152;
#pragma unroll
  for (int kp = 0; kp < 9; ++kp) {
    float acc = 0.f;
#pragma unroll
    for (int p = 0; p < 9; ++p) acc += wg[rowOf[p] * 9 + kp] * ys[p][c];
    unsafeAtomicAdd(&ob[kp * 128 + c], env * acc);
  }
}

// ---------------- launch ----------------
extern "C" void kernel_launch(void* const* d_in, const int* in_sizes, int n_in,
                              void* d_out, int out_size, void* d_ws, size_t ws_size,
                              hipStream_t stream) {
  const float* node_feats   = (const float*)d_in[0];
  const float* edge_embeds  = (const float*)d_in[1];
  const int*   senders      = (const int*)d_in[2];
  const int*   receivers    = (const int*)d_in[3];
  const float* wigner       = (const float*)d_in[4];
  const float* edge_env     = (const float*)d_in[5];
  const float* rad_w1       = (const float*)d_in[6];
  const float* rad_b1       = (const float*)d_in[7];
  const float* rad_w2       = (const float*)d_in[8];
  const float* rad_b2       = (const float*)d_in[9];
  const float* conv1_w0     = (const float*)d_in[10];
  const float* conv1_b0     = (const float*)d_in[11];
  const float* conv1_wr1    = (const float*)d_in[12];
  const float* conv1_wi1    = (const float*)d_in[13];
  const float* conv1_wr2    = (const float*)d_in[14];
  const float* conv1_wi2    = (const float*)d_in[15];
  const float* conv2_w0     = (const float*)d_in[16];
  const float* conv2_b0     = (const float*)d_in[17];
  const float* conv2_wr1    = (const float*)d_in[18];
  const float* conv2_wi1    = (const float*)d_in[19];
  const float* conv2_wr2    = (const float*)d_in[20];
  const float* conv2_wi2    = (const float*)d_in[21];

  char* ws = (char*)d_ws;
  bf16_t* wb    = (bf16_t*)ws;
  bf16_t* bufA  = (bf16_t*)(ws + B0_OFF);
  bf16_t* bufY1 = (bf16_t*)(ws + B1_OFF);
  bf16_t* bufA2 = (bf16_t*)(ws + B2_OFF);
  bf16_t* bufY2 = (bf16_t*)(ws + B3_OFF);
  float* out = (float*)d_out;

  const long outN = (long)out_size;
  zero_kernel<<<dim3((unsigned)((outN + 255) / 256)), 256, 0, stream>>>(out, outN);

  pack_kernel<<<dim3(1024), 256, 0, stream>>>(rad_w1, rad_w2, conv1_w0, conv1_wr1, conv1_wi1,
                                              conv1_wr2, conv1_wi2, conv2_w0, conv2_wr1,
                                              conv2_wi1, conv2_wr2, conv2_wi2, wb);

  prep_kernel<<<dim3(NUM_E / 16, 2), 128, 0, stream>>>(node_feats, edge_embeds, senders,
                                                       receivers, wigner, wb, rad_b1, rad_b2,
                                                       bufA);

  // ---- SO2 conv 1 ----
  gemm_bf16_kernel<<<dim3(NUM_E / 128, 640 / 64), 256, 0, stream>>>(
      bufA, 2304, 0, wb + OFF_W0C1, 640, conv1_b0, bufY1, 1408, 0, 768);
  gemm_bf16_kernel<<<dim3(NUM_E / 128, 512 / 64), 256, 0, stream>>>(
      bufA, 2304, 768, wb + OFF_W1C1, 512, nullptr, bufY1, 1408, 640, 1024);
  gemm_bf16_kernel<<<dim3(NUM_E / 128, 256 / 64), 256, 0, stream>>>(
      bufA, 2304, 1792, wb + OFF_W2C1, 256, nullptr, bufY1, 1408, 1152, 512);

  gate_kernel<<<dim3(NUM_E), 128, 0, stream>>>(bufY1, bufA2);

  // ---- SO2 conv 2 ----
  gemm_bf16_kernel<<<dim3(NUM_E / 128, 384 / 64), 256, 0, stream>>>(
      bufA2, 1152, 0, wb + OFF_W0C2, 384, conv2_b0, bufY2, 1152, 0, 384);
  gemm_bf16_kernel<<<dim3(NUM_E / 128, 512 / 64), 256, 0, stream>>>(
      bufA2, 1152, 384, wb + OFF_W1C2, 512, nullptr, bufY2, 1152, 384, 512);
  gemm_bf16_kernel<<<dim3(NUM_E / 128, 256 / 64), 256, 0, stream>>>(
      bufA2, 1152, 896, wb + OFF_W2C2, 256, nullptr, bufY2, 1152, 896, 256);

  scatter_kernel<<<dim3(NUM_E), 128, 0, stream>>>(bufY2, wigner, edge_env, receivers, out);
}
